// MultiHeadSelfAttention_89464168775741
// MI455X (gfx1250) — matrix operand order repaired and checked by reading, not compiled, not run on hardware
//
#include <hip/hip_runtime.h>

// ---------------------------------------------------------------------------
// Types for CDNA5 WMMA (wave32): A/B = 16 bf16 per lane, C/D = 8 f32 per lane
// ---------------------------------------------------------------------------
typedef __attribute__((ext_vector_type(16))) __bf16 v16bf;
typedef __attribute__((ext_vector_type(8)))  __bf16 v8bf;
typedef __attribute__((ext_vector_type(4)))  __bf16 v4bf;
typedef __attribute__((ext_vector_type(8)))  float  v8f;
typedef __attribute__((ext_vector_type(4)))  int    v4i;

// ---------------------------------------------------------------------------
// Async global->LDS staging (CDNA5 GLOBAL_LOAD_ASYNC_TO_LDS_B128, ASYNCcnt).
// Builtin signature (from probe): (v4i __device__*, v4i __shared__*, i32, i32)
// ---------------------------------------------------------------------------
#if defined(__has_builtin)
#if __has_builtin(__builtin_amdgcn_global_load_async_to_lds_b128) && \
    __has_builtin(__builtin_amdgcn_s_wait_asynccnt)
#define ASYNC_LDS 1
#endif
#endif
#ifndef ASYNC_LDS
#define ASYNC_LDS 0
#endif

static __device__ __forceinline__ void stage16(__bf16* lds_dst, const __bf16* gsrc) {
#if ASYNC_LDS
  __builtin_amdgcn_global_load_async_to_lds_b128(
      (__attribute__((address_space(1))) v4i*)gsrc,
      (__attribute__((address_space(3))) v4i*)lds_dst, 0, 0);
#else
  *reinterpret_cast<v8bf*>(lds_dst) = *reinterpret_cast<const v8bf*>(gsrc);
#endif
}

static __device__ __forceinline__ void async_fence() {
#if ASYNC_LDS
  __builtin_amdgcn_s_wait_asynccnt(0);
#endif
}

static __device__ __forceinline__ v8f wmma_bf16(v16bf a, v16bf b, v8f c) {
  // D(16x16 f32) = A(16x32 bf16) * B(32x16 bf16) + C
  return __builtin_amdgcn_wmma_f32_16x16x32_bf16(false, a, false, b, (short)0, c,
                                                 false, false);
}

static __device__ __forceinline__ v8f zero8() {
  v8f z = {0.f, 0.f, 0.f, 0.f, 0.f, 0.f, 0.f, 0.f};
  return z;
}

static __device__ __forceinline__ v16bf load2x8(const __bf16* p0, const __bf16* p1) {
  v8bf lo = *reinterpret_cast<const v8bf*>(p0);
  v8bf hi = *reinterpret_cast<const v8bf*>(p1);
  v16bf r;
#pragma unroll
  for (int i = 0; i < 8; ++i) { r[i] = lo[i]; r[i + 8] = hi[i]; }
  return r;
}

// A-matrix fragment (16x32 bf16, MxK). Lane l: row m=l&15, half g=l>>4;
// elements 0..7 -> k = 8g+e, elements 8..15 -> k = 16+8g+(e-8).
static __device__ __forceinline__ v16bf load_a_frag(const __bf16* base, int stride,
                                                    int lane) {
  int m = lane & 15, g = lane >> 4;
  const __bf16* p = base + m * stride + 8 * g;
  return load2x8(p, p + 16);
}

// B-matrix fragment (32x16 bf16, KxN) from N-major storage BT[n][k]:
// lane l: column n=l&15, half g=l>>4; elements 0..7 -> k = 8g+e, 8..15 -> k = 16+8g+(e-8).
static __device__ __forceinline__ v16bf load_b_frag_t(const __bf16* baseT, int stride,
                                                      int lane) {
  int n = lane & 15, g = lane >> 4;
  const __bf16* p = baseT + n * stride + 8 * g;
  return load2x8(p, p + 16);
}

// ---------------------------------------------------------------------------
// permlane16-based xor butterfly within each 16-lane half (VALU, no LDS pipe)
// ---------------------------------------------------------------------------
static constexpr unsigned psel(int m, int base) {
  unsigned v = 0;
  for (int i = 0; i < 8; ++i) v |= (unsigned)(((base + i) ^ m) & 0xF) << (4 * i);
  return v;
}

template <int XM>
static __device__ __forceinline__ float xor16f(float v) {
  constexpr unsigned s0 = psel(XM, 0);
  constexpr unsigned s1 = psel(XM, 8);
  int r = __builtin_amdgcn_permlane16(__float_as_int(v), __float_as_int(v),
                                      (int)s0, (int)s1, false, false);
  return __int_as_float(r);
}

static __device__ __forceinline__ float rowmax16(float v) {
  v = fmaxf(v, xor16f<1>(v));
  v = fmaxf(v, xor16f<2>(v));
  v = fmaxf(v, xor16f<4>(v));
  v = fmaxf(v, xor16f<8>(v));
  return v;
}
static __device__ __forceinline__ float rowsum16(float v) {
  v += xor16f<1>(v);
  v += xor16f<2>(v);
  v += xor16f<4>(v);
  v += xor16f<8>(v);
  return v;
}

// ---------------------------------------------------------------------------
// fp32 -> bf16 conversion (vectorized)
// ---------------------------------------------------------------------------
__global__ void __launch_bounds__(256)
cvt_f32_bf16(const float4* __restrict__ in, __bf16* __restrict__ out, int n4) {
  int i = blockIdx.x * blockDim.x + threadIdx.x;
  int stride = gridDim.x * blockDim.x;
  for (; i < n4; i += stride) {
    float4 v = in[i];
    v4bf o;
    o[0] = (__bf16)v.x; o[1] = (__bf16)v.y; o[2] = (__bf16)v.z; o[3] = (__bf16)v.w;
    *reinterpret_cast<v4bf*>(out + 4 * (size_t)i) = o;
  }
}

// ---------------------------------------------------------------------------
// fp32 [K,N] row-major -> bf16 [N,K] (n-major), LDS-tiled 64x64 transpose.
// ---------------------------------------------------------------------------
__global__ void __launch_bounds__(256)
cvt_transpose(const float* __restrict__ in, __bf16* __restrict__ out, int K, int N) {
  __shared__ float tile[64][65];
  const int k0 = blockIdx.y * 64, n0 = blockIdx.x * 64;
  const int tid = threadIdx.x;
  {
    int r = tid >> 4, c = (tid & 15) * 4;
#pragma unroll
    for (int it = 0; it < 4; ++it) {
      int kk = r + it * 16;
      float4 v = *reinterpret_cast<const float4*>(in + (size_t)(k0 + kk) * N + n0 + c);
      tile[kk][c + 0] = v.x; tile[kk][c + 1] = v.y;
      tile[kk][c + 2] = v.z; tile[kk][c + 3] = v.w;
    }
  }
  __syncthreads();
  {
    int nn = tid >> 2, kc = (tid & 3) * 16;
    v8bf o0, o1;
#pragma unroll
    for (int i = 0; i < 8; ++i) o0[i] = (__bf16)tile[kc + i][nn];
#pragma unroll
    for (int i = 0; i < 8; ++i) o1[i] = (__bf16)tile[kc + 8 + i][nn];
    __bf16* dst = out + (size_t)(n0 + nn) * K + k0 + kc;
    *reinterpret_cast<v8bf*>(dst) = o0;
    *reinterpret_cast<v8bf*>(dst + 8) = o1;
  }
}

// ---------------------------------------------------------------------------
// Tiled bf16 GEMM, f32 accumulation: out[M,N] = A[M,K] @ BT^T + bias[N]
// Block 128x128, 128 threads = 4 waves; wave = 64x64 subtile (16 WMMA/K-step).
// Async double-buffered LDS pipeline: one barrier per K-step, next tile lands
// while WMMAs run on the current one.
// ---------------------------------------------------------------------------
template <typename OUT_T, bool SPLIT_V>
__global__ void __launch_bounds__(128)
gemm_bf16(const __bf16* __restrict__ A, const __bf16* __restrict__ BT,
          const float* __restrict__ bias, OUT_T* __restrict__ out,
          __bf16* __restrict__ vt, int M, int N, int K) {
  constexpr int AS = 40;  // padded element stride: 80B rows, 16B aligned
  constexpr int BS = 40;
  __shared__ __bf16 As[2][128 * AS];  // [m][k] row-major, 32 k per step
  __shared__ __bf16 Bs[2][128 * BS];  // [n][k] n-major

  const int tid = threadIdx.x;
  const int lane = tid & 31;
  const int w = tid >> 5;          // 0..3
  const int wm = (w >> 1) * 64;
  const int wn = (w & 1) * 64;
  const int bm = blockIdx.y * 128;
  const int bn = blockIdx.x * 128;

  const __bf16* asrc0 = A + (size_t)(bm + tid) * K;
  const __bf16* bsrc0 = BT + (size_t)(bn + tid) * K;

  auto stage_tile = [&](int buf, int k0) {
#pragma unroll
    for (int i = 0; i < 4; ++i)
      stage16(&As[buf][tid * AS + 8 * i], asrc0 + k0 + 8 * i);
#pragma unroll
    for (int i = 0; i < 4; ++i)
      stage16(&Bs[buf][tid * BS + 8 * i], bsrc0 + k0 + 8 * i);
  };

  v8f acc[4][4];
#pragma unroll
  for (int i = 0; i < 4; ++i)
#pragma unroll
    for (int j = 0; j < 4; ++j) acc[i][j] = zero8();

  stage_tile(0, 0);
  int buf = 0;
  for (int k0 = 0; k0 < K; k0 += 32) {
    async_fence();        // this wave's staged tile has landed
    __syncthreads();      // everyone's tile visible; prev buffer free
    if (k0 + 32 < K) stage_tile(buf ^ 1, k0 + 32);  // overlaps with compute

    v16bf af[4];
#pragma unroll
    for (int i = 0; i < 4; ++i)
      af[i] = load_a_frag(&As[buf][(wm + i * 16) * AS], AS, lane);
#pragma unroll
    for (int j = 0; j < 4; ++j) {
      v16bf bj = load_b_frag_t(&Bs[buf][(wn + j * 16) * BS], BS, lane);
#pragma unroll
      for (int i = 0; i < 4; ++i) acc[i][j] = wmma_bf16(af[i], bj, acc[i][j]);
    }
    buf ^= 1;
  }

  // --- epilogue (C layout: lane holds col n=lane&15, rows r+8g) ---
  const int nl = lane & 15, g = lane >> 4;
#pragma unroll
  for (int j = 0; j < 4; ++j) {
    int col = bn + wn + j * 16 + nl;
    float bv = bias[col];
    float qscale = (SPLIT_V && col < 1024) ? 0.125f : 1.0f;
#pragma unroll
    for (int i = 0; i < 4; ++i) {
      int row0 = bm + wm + i * 16 + 8 * g;
      if (SPLIT_V && col >= 2048) {
        // V third -> vt[(b*H+h)*D+d][t]; 8 consecutive t = one 16B store
        int b = row0 >> 11, t0 = row0 & 2047;
        int hd = col - 2048;  // h*64 + d
        v8bf pk;
#pragma unroll
        for (int r = 0; r < 8; ++r) pk[r] = (__bf16)(acc[i][j][r] + bv);
        *reinterpret_cast<v8bf*>(vt + ((size_t)(b * 1024 + hd) << 11) + t0) = pk;
      } else {
#pragma unroll
        for (int r = 0; r < 8; ++r) {
          float v = (acc[i][j][r] + bv) * qscale;
          out[(size_t)(row0 + r) * N + col] = (OUT_T)v;
        }
      }
    }
  }
}

// ---------------------------------------------------------------------------
// Flash attention (causal). qkv: bf16 [B*T,3C] (Q pre-scaled by 1/8), vt: bf16
// [B,H,D,T], o: bf16 [B*T,C].  Grid: (T/128, H, B). Block: 256 = 8 waves; wave
// w owns 16 query rows; 32-key tiles staged async + double-buffered.
// ---------------------------------------------------------------------------
__global__ void __launch_bounds__(256)
attn_kernel(const __bf16* __restrict__ qkv, const __bf16* __restrict__ vt,
            __bf16* __restrict__ o) {
  constexpr int T = 2048, C = 1024, D = 64, H = 16;
  constexpr int RS = 3 * C;   // qkv row stride
  constexpr int KS = 72;      // K tile stride (d padded 64->72; 144B rows)
  constexpr int VS = 40;      // V^T tile stride (key padded 32->40; 80B rows)
  constexpr int PS = 40;      // P strip stride
  __shared__ __bf16 Kt[2][32 * KS];   // [key][d] row-major
  __shared__ __bf16 Vt[2][D * VS];    // [d][key]
  __shared__ __bf16 Pt[8][16 * PS];   // per-wave probability strip

  const int tid = threadIdx.x;
  const int lane = tid & 31;
  const int w = tid >> 5;             // 0..7
  const int b = blockIdx.z, h = blockIdx.y;
  const int qb = blockIdx.x * 128;
  const int qw = qb + w * 16;         // this wave's first query row
  const int hcol = h * D;
  const int nl = lane & 15, g = lane >> 4;

  const __bf16* qkv_b = qkv + (size_t)b * T * RS;
  const __bf16* vth = vt + ((size_t)(b * H + h) * D) * T;

  // Per-thread staging sources: one 16B chunk of K, one of V^T
  const int skey = tid >> 3, sc = (tid & 7) * 8;   // K: [key][d-chunk]
  const int sd = tid >> 2, svh = (tid & 3) * 8;    // V: [d][key-chunk]
  const __bf16* ksrc0 = qkv_b + (size_t)skey * RS + C + hcol + sc;
  const __bf16* vsrc0 = vth + (size_t)sd * T + svh;

  auto stage_kv = [&](int buf, int kt) {
    stage16(&Kt[buf][skey * KS + sc], ksrc0 + (size_t)kt * RS);
    stage16(&Vt[buf][sd * VS + svh], vsrc0 + kt);
  };

  // Q fragments (already scaled by 1/sqrt(D) in GEMM1 epilogue)
  const __bf16* qbase = qkv_b + (size_t)qw * RS + hcol;
  v16bf qa0 = load_a_frag(qbase, RS, lane);
  v16bf qa1 = load_a_frag(qbase + 32, RS, lane);

  v8f o_acc[4];
#pragma unroll
  for (int j = 0; j < 4; ++j) o_acc[j] = zero8();
  float rmax[8], rsum[8];
#pragma unroll
  for (int r = 0; r < 8; ++r) { rmax[r] = -1.0e30f; rsum[r] = 0.f; }

  const int kend = qb + 128;
  stage_kv(0, 0);
  int buf = 0;
  for (int kt = 0; kt < kend; kt += 32) {
    async_fence();        // this wave's K/V chunks landed
    __syncthreads();      // all staged data visible; prev buffer free
    if (kt + 32 < kend) stage_kv(buf ^ 1, kt + 32);  // overlaps with compute

    // Skip tiles entirely above this wave's causal diagonal.
    if (kt <= qw + 15) {
      // --- S = Q @ K^T for two 16-key subtiles ---
      v8f s[2];
#pragma unroll
      for (int t = 0; t < 2; ++t) {
        v16bf kb0 = load_b_frag_t(&Kt[buf][(t * 16) * KS + 0], KS, lane);
        v16bf kb1 = load_b_frag_t(&Kt[buf][(t * 16) * KS + 32], KS, lane);
        v8f z = zero8();
        z = wmma_bf16(qa0, kb0, z);
        z = wmma_bf16(qa1, kb1, z);
        s[t] = z;
      }
      // --- causal mask only near the diagonal ---
      if (kt + 31 > qw) {
#pragma unroll
        for (int t = 0; t < 2; ++t) {
          int rel = kt + t * 16 + nl - qw - 8 * g;  // masked iff rel > r
#pragma unroll
          for (int r = 0; r < 8; ++r)
            s[t][r] = (rel <= r) ? s[t][r] : -1.0e30f;
        }
      }
      // --- online softmax (permlane16 butterflies, VALU only) ---
      float corr[8];
#pragma unroll
      for (int r = 0; r < 8; ++r) {
        float v = rowmax16(fmaxf(s[0][r], s[1][r]));
        float nm = fmaxf(rmax[r], v);
        corr[r] = __expf(rmax[r] - nm);
        rmax[r] = nm;
        float p0 = __expf(s[0][r] - nm);
        float p1 = __expf(s[1][r] - nm);
        s[0][r] = p0;
        s[1][r] = p1;
        rsum[r] = rsum[r] * corr[r] + rowsum16(p0 + p1);
      }
#pragma unroll
      for (int j = 0; j < 4; ++j)
#pragma unroll
        for (int r = 0; r < 8; ++r) o_acc[j][r] *= corr[r];

      // --- P (16x32) to LDS, re-read as one A fragment ---
      __bf16* P = Pt[w];
#pragma unroll
      for (int t = 0; t < 2; ++t)
#pragma unroll
        for (int r = 0; r < 8; ++r)
          P[(r + 8 * g) * PS + t * 16 + nl] = (__bf16)s[t][r];
      v16bf pa = load_a_frag(P, PS, lane);  // LDS in-order within wave

      // --- O += P @ V ---
#pragma unroll
      for (int j = 0; j < 4; ++j) {
        v16bf vb = load_b_frag_t(&Vt[buf][(j * 16) * VS], VS, lane);
        o_acc[j] = wmma_bf16(pa, vb, o_acc[j]);
      }
    }
    buf ^= 1;
  }

  // --- finalize: O /= rowsum, store bf16 [B*T, C] ---
  float inv[8];
#pragma unroll
  for (int r = 0; r < 8; ++r) inv[r] = 1.0f / rsum[r];
#pragma unroll
  for (int j = 0; j < 4; ++j) {
    int d = j * 16 + nl;
#pragma unroll
    for (int r = 0; r < 8; ++r) {
      int row = qw + r + 8 * g;
      float val = o_acc[j][r] * inv[r];
      o[(size_t)(b * T + row) * C + hcol + d] = (__bf16)val;
    }
  }
}

// ---------------------------------------------------------------------------
// Host-side launcher
// ---------------------------------------------------------------------------
extern "C" void kernel_launch(void* const* d_in, const int* in_sizes, int n_in,
                              void* d_out, int out_size, void* d_ws, size_t ws_size,
                              hipStream_t stream) {
  (void)in_sizes; (void)n_in; (void)out_size; (void)ws_size;
  constexpr int B = 2, T = 2048, C = 1024, H = 16;
  constexpr int M = B * T;        // 4096
  constexpr int N1 = 3 * C;       // 3072

  const float* x     = (const float*)d_in[0];
  const float* w_qkv = (const float*)d_in[1];
  const float* b_qkv = (const float*)d_in[2];
  const float* w_out = (const float*)d_in[3];
  const float* b_out = (const float*)d_in[4];
  float* out = (float*)d_out;

  // Workspace layout (bytes)
  char* ws = (char*)d_ws;
  __bf16* xb     = (__bf16*)(ws);                    //  8 MiB [4096,1024]
  __bf16* wqkvT  = (__bf16*)(ws + 8388608);          //  6 MiB [3072,1024] n-major
  __bf16* woutT  = (__bf16*)(ws + 14680064);         //  2 MiB [1024,1024] n-major
  __bf16* qkvb   = (__bf16*)(ws + 16777216);         // 24 MiB [4096,3072] (Q,K)
  __bf16* vtb    = (__bf16*)(ws + 41943040);         //  8 MiB [B,H,D,T]
  __bf16* ob     = (__bf16*)(ws + 50331648);         //  8 MiB [4096,1024]
  // total 56 MiB

  // 1) input conversions (weights transposed to n-major)
  cvt_f32_bf16<<<1024, 256, 0, stream>>>((const float4*)x, xb, M * C / 4);
  cvt_transpose<<<dim3(N1 / 64, C / 64), 256, 0, stream>>>(w_qkv, wqkvT, C, N1);
  cvt_transpose<<<dim3(C / 64, C / 64), 256, 0, stream>>>(w_out, woutT, C, C);

  // 2) qkv = x @ w_qkv + b_qkv ; Q scaled by 1/8 ; V scattered to [B,H,D,T]
  gemm_bf16<__bf16, true><<<dim3(N1 / 128, M / 128), 128, 0, stream>>>(
      xb, wqkvT, b_qkv, qkvb, vtb, M, N1, C);

  // 3) causal flash attention
  attn_kernel<<<dim3(T / 128, H, B), 256, 0, stream>>>(qkvb, vtb, ob);

  // 4) out = o @ w_out + b_out (f32)
  gemm_bf16<float, false><<<dim3(C / 128, M / 128), 128, 0, stream>>>(
      ob, woutT, b_out, out, nullptr, M, C, C);
}
